// REV_BETA_VAE_43568148251308
// MI455X (gfx1250) — matrix operand order repaired and checked by reading, not compiled, not run on hardware
//
#include <hip/hip_runtime.h>
#include <hip/hip_bf16.h>
#include <cstdint>
#include <cstddef>

typedef __attribute__((ext_vector_type(16))) _Float16 v16h;
typedef __attribute__((ext_vector_type(8)))  float    v8f;

#if defined(__has_builtin)
#  if __has_builtin(__builtin_amdgcn_global_load_async_to_lds_b64) && \
      __has_builtin(__builtin_amdgcn_s_wait_asynccnt)
#    define HAVE_ASYNC_LDS 1
#  endif
#endif
#ifndef HAVE_ASYNC_LDS
#  define HAVE_ASYNC_LDS 0
#endif

#if HAVE_ASYNC_LDS
typedef int v2i_t __attribute__((ext_vector_type(2)));
typedef __attribute__((address_space(1))) v2i_t* as1_v2i_p;   // global
typedef __attribute__((address_space(3))) v2i_t* as3_v2i_p;   // LDS
#endif

// ---------------------------------------------------------------------------
// Fully-templated implicit-im2col WMMA GEMM.
// 128 threads = 4 waves; block tile 64(M) x 16(N); K stepped by 32 via LDS.
// OPMODE: 0 = conv/fc, 1 = transposed conv. ACT: 0 none, 1 relu, 2 sigmoid.
// B (weight) tile staged with async global->LDS DMA when available.
// ---------------------------------------------------------------------------
template<int OPMODE, int ACT, int CIN, int HIN, int WIN,
         int NREAL, int NPAD, int HOUT, int WOUT,
         int KH, int KW, int STRIDE, int PAD, int KREAL, int KPAD>
__global__ __launch_bounds__(128)
void vae_wmma_gemm(const _Float16* __restrict__ act,
                   const _Float16* __restrict__ wgt,   // packed [NPAD][KPAD] f16
                   const float*    __restrict__ bias,  // [>=NREAL]
                   _Float16*       __restrict__ outH,  // f16 NCHW (or null)
                   float*          __restrict__ outF,  // f32 NCHW (or null)
                   int M)
{
    constexpr int BSTRIDE = HAVE_ASYNC_LDS ? 16 : 17;  // dwords per B row
    __shared__ unsigned int sA[64 * 17];   // 64 rows x 32 f16 (rows padded)
    __shared__ unsigned int sB[16 * 17];   // 16 cols x 32 f16 (transposed)

    constexpr int HW  = HOUT * WOUT;
    constexpr int KHW = KH * KW;

    const int tid  = threadIdx.x;
    const int wv   = tid >> 5;
    const int lane = tid & 31;
    const int m0   = blockIdx.x * 64;
    const int n0   = blockIdx.y * 16;

    // ---- per-thread A-row decomposition (hoisted out of the K loop) ----
    const int  arow   = tid >> 1;           // 0..63
    const int  ahalf  = (tid & 1) * 16;     // 0 or 16
    const int  m_fill = m0 + arow;
    const bool mvalid = (m_fill < M);
    const int  b_f    = m_fill / HW;        // const divisor -> magic mul
    const int  rem_f  = m_fill - b_f * HW;
    const int  oy_f   = rem_f / WOUT;
    const int  ox_f   = rem_f - oy_f * WOUT;
    const _Float16* __restrict__ actB = act + (size_t)b_f * (CIN * HIN * WIN);

    // B-tile fill coordinates (one 8-byte chunk per thread)
    const int bn  = tid >> 3;               // 0..15
    const int bk0 = (tid & 7) * 4;          // 0,4,...,28 (f16 units)

    v8f c = {0.f, 0.f, 0.f, 0.f, 0.f, 0.f, 0.f, 0.f};

    for (int kk = 0; kk < KPAD; kk += 32) {
        __syncthreads();

#if HAVE_ASYNC_LDS
        // ---- async DMA the B tile (global -> LDS), overlaps the A gather ----
        {
            const _Float16* src = &wgt[(size_t)(n0 + bn) * KPAD + kk + bk0];
            _Float16* dst = (_Float16*)&sB[bn * BSTRIDE] + bk0;
            __builtin_amdgcn_global_load_async_to_lds_b64(
                (as1_v2i_p)(void*)src, (as3_v2i_p)(void*)dst, 0, 0);
        }
#endif
        // ---- stage A tile: implicit im2col gather, 16 f16 per thread ----
        {
            _Float16* rowp = (_Float16*)&sA[arow * 17];
            #pragma unroll
            for (int i = 0; i < 16; ++i) {
                const int kg = kk + ahalf + i;
                _Float16 v = (_Float16)0.0f;
                if (mvalid && (KREAL == KPAD || kg < KREAL)) {
                    const int ci = kg / KHW;           // compile-time divisor
                    const int r  = kg - ci * KHW;
                    const int ky = r / KW;
                    const int kx = r - ky * KW;
                    if (OPMODE == 0) {                 // conv / fc
                        const int iy = oy_f * STRIDE - PAD + ky;
                        const int ix = ox_f * STRIDE - PAD + kx;
                        if (iy >= 0 && iy < HIN && ix >= 0 && ix < WIN)
                            v = actB[(ci * HIN + iy) * WIN + ix];
                    } else {                           // transposed conv
                        const int ty = oy_f + PAD - ky;
                        const int tx = ox_f + PAD - kx;
                        if (ty >= 0 && tx >= 0 &&
                            (ty % STRIDE) == 0 && (tx % STRIDE) == 0) {
                            const int iy = ty / STRIDE;
                            const int ix = tx / STRIDE;
                            if (iy < HIN && ix < WIN)
                                v = actB[(ci * HIN + iy) * WIN + ix];
                        }
                    }
                }
                rowp[ahalf + i] = v;
            }
        }
#if HAVE_ASYNC_LDS
        __builtin_amdgcn_s_wait_asynccnt(0);
#else
        // ---- fallback: direct-copy B tile, 4 contiguous f16/thread ----
        {
            _Float16* rowp = (_Float16*)&sB[bn * BSTRIDE];
            const _Float16* src = &wgt[(size_t)(n0 + bn) * KPAD + kk + bk0];
            #pragma unroll
            for (int j = 0; j < 4; ++j) rowp[bk0 + j] = src[j];
        }
        // prefetch next weight slab (global_prefetch_b8)
        if (kk + 32 < KPAD && tid < 16)
            __builtin_prefetch(&wgt[(size_t)(n0 + tid) * KPAD + kk + 32], 0, 0);
#endif
        __syncthreads();

        // ---- fragments per ISA 7.12.2 layouts ----
        union { unsigned int u[8]; v16h h; } fa, fb;
        {   // A 16x32 f16
            const int mr  = wv * 16 + (lane & 15);
            const int kh2 = (lane >> 4) * 4;
            #pragma unroll
            for (int v = 0; v < 4; ++v) {
                fa.u[v]     = sA[mr * 17 + kh2 + v];
                fa.u[v + 4] = sA[mr * 17 + 8 + kh2 + v];
            }
        }
        {   // B 32x16 f16
            const int nr = lane & 15;
            const int kb = (lane >> 4) * 4;
            #pragma unroll
            for (int v = 0; v < 4; ++v) {
                fb.u[v]     = sB[nr * BSTRIDE + kb + v];
                fb.u[v + 4] = sB[nr * BSTRIDE + 8 + kb + v];
            }
        }
        c = __builtin_amdgcn_wmma_f32_16x16x32_f16(false, fa.h, false, fb.h,
                                                   (short)0, c, false, false);
    }

    // ---- epilogue: bias + activation, scatter to NCHW ----
    // The 8 rows per lane are consecutive m values: decompose once, then
    // carry-increment ox -> oy -> b instead of re-dividing per element.
    const int n_out = n0 + (lane & 15);
    const int m_base = m0 + wv * 16 + 8 * (lane >> 4);
    int b_o   = m_base / HW;
    int rem_o = m_base - b_o * HW;
    int oy_o  = rem_o / WOUT;
    int ox_o  = rem_o - oy_o * WOUT;
    #pragma unroll
    for (int r = 0; r < 8; ++r) {
        const int m_out = m_base + r;
        if (m_out < M && n_out < NREAL) {
            float v = c[r] + bias[n_out];
            if      (ACT == 1) v = v > 0.f ? v : 0.f;
            else if (ACT == 2) v = 1.f / (1.f + __expf(-v));
            const size_t off =
                (size_t)(((b_o * NREAL + n_out) * HOUT + oy_o)) * WOUT + ox_o;
            if (outH) outH[off] = (_Float16)v;
            if (outF) outF[off] = v;
        }
        // carry-chain increment of (b_o, oy_o, ox_o)
        ++ox_o;
        if (ox_o == WOUT) {
            ox_o = 0;
            ++oy_o;
            if (oy_o == HOUT) { oy_o = 0; ++b_o; }
        }
    }
}

// ---------------------------------------------------------------------------
// Weight repack: f32 source -> zero-padded f16 [Npad][Kpad] (N-major)
// mode 0: conv/fc  src[N][Cin][kh][kw] ; mode 1: deconv src[Cin][Cout][kh][kw]
// ---------------------------------------------------------------------------
__global__ void vae_pack_w(const float* __restrict__ src, _Float16* __restrict__ dst,
                           int Nreal, int Npad, int Kreal, int Kpad,
                           int CinOrCout, int kh, int kw, int mode)
{
    int i = blockIdx.x * blockDim.x + threadIdx.x;
    if (i >= Npad * Kpad) return;
    int n = i / Kpad, k = i - n * Kpad;
    float v = 0.f;
    if (n < Nreal && k < Kreal) {
        int khw = kh * kw;
        int ci  = k / khw;
        int r   = k - ci * khw;
        int ky  = r / kw, kx = r - ky * kw;
        int idx;
        if (mode == 0) idx = ((n * CinOrCout + ci) * kh + ky) * kw + kx;
        else           idx = ((ci * CinOrCout + n) * kh + ky) * kw + kx;
        v = src[idx];
    }
    dst[(size_t)n * Kpad + k] = (_Float16)v;
}

__global__ void vae_f32_to_f16(const float* __restrict__ src,
                               _Float16* __restrict__ dst, int n)
{
    int i = blockIdx.x * blockDim.x + threadIdx.x;
    if (i < n) dst[i] = (_Float16)src[i];
}

__global__ void vae_head_bias(const float* __restrict__ bmu,
                              const float* __restrict__ bls, float* __restrict__ dst)
{
    int i = threadIdx.x;   // 48 lanes used
    if (i < 48) dst[i] = (i < 20) ? bmu[i] : (i < 40 ? bls[i - 20] : 0.f);
}

// ---------------------------------------------------------------------------
// Latent: emit mu/logvar, run 100-step KLD gradient-ascent, build z [B,20] f16
// ---------------------------------------------------------------------------
__global__ void vae_latent(const float* __restrict__ headF,  // [B][40]
                           const float* __restrict__ eps,    // [B][20]
                           float* __restrict__ out_mu, float* __restrict__ out_ls,
                           _Float16* __restrict__ zlat, int Bn)
{
    int i = blockIdx.x * blockDim.x + threadIdx.x;
    if (i >= Bn * 20) return;
    int b = i / 20, j = i - b * 20;
    float mu = headF[b * 40 + j];
    float ls = headF[b * 40 + 20 + j];
    out_mu[i] = mu;
    out_ls[i] = ls;
    float m = mu, s = ls;
    #pragma unroll 4
    for (int t = 0; t < 100; ++t) {
        m += 0.1f * m;
        s += 0.05f * (__expf(s) - 1.0f);
    }
    zlat[i] = (_Float16)(eps[i] * __expf(0.5f * s) + m);
}

// ---------------------------------------------------------------------------
// Host side
// ---------------------------------------------------------------------------
static inline void launch_pack(const float* src, _Float16* dst, int Nr, int Np,
                               int Kr, int Kp, int cc, int kh, int kw, int mode,
                               hipStream_t s) {
    int n = Np * Kp;
    vae_pack_w<<<(n + 255) / 256, 256, 0, s>>>(src, dst, Nr, Np, Kr, Kp, cc, kh, kw, mode);
}

extern "C" void kernel_launch(void* const* d_in, const int* in_sizes, int n_in,
                              void* d_out, int out_size, void* d_ws, size_t ws_size,
                              hipStream_t stream)
{
    (void)in_sizes; (void)n_in; (void)out_size; (void)ws_size;
    const int B = 4096;

    const float* x     = (const float*)d_in[0];
    const float* eps   = (const float*)d_in[1];
    const float* w_c1  = (const float*)d_in[2];  const float* b_c1 = (const float*)d_in[3];
    const float* w_c2  = (const float*)d_in[4];  const float* b_c2 = (const float*)d_in[5];
    const float* w_c3  = (const float*)d_in[6];  const float* b_c3 = (const float*)d_in[7];
    const float* w_fc1 = (const float*)d_in[8];  const float* b_fc1= (const float*)d_in[9];
    const float* w_mu  = (const float*)d_in[10]; const float* b_mu = (const float*)d_in[11];
    const float* w_ls  = (const float*)d_in[12]; const float* b_ls = (const float*)d_in[13];
    const float* w_fc2 = (const float*)d_in[14]; const float* b_fc2= (const float*)d_in[15];
    const float* w_fc3 = (const float*)d_in[16]; const float* b_fc3= (const float*)d_in[17];
    const float* w_d1  = (const float*)d_in[18]; const float* b_d1 = (const float*)d_in[19];
    const float* w_d2  = (const float*)d_in[20]; const float* b_d2 = (const float*)d_in[21];
    const float* w_d3  = (const float*)d_in[22]; const float* b_d3 = (const float*)d_in[23];

    char* ws = (char*)d_ws;
    const size_t OFF_HB    = 1536ull * 1024;
    const size_t OFF_XIN   = 2ull  << 20;        // x f16 [B,1,28,28]
    const size_t OFF_A1    = 10ull << 20;        // [B,32,14,14] f16  (reused as E2)
    const size_t OFF_A2    = 62ull << 20;        // [B,32,7,7]  f16  (reused as E1)
    const size_t OFF_A3    = 76ull << 20;        // [B,64,4,4]  f16  (reused as D2)
    const size_t OFF_H     = 86ull << 20;        // [B,256]     f16  (reused as D1)
    const size_t OFF_HEADF = 89ull << 20;        // [B,40]      f32
    const size_t OFF_Z     = 90ull << 20;        // [B,20]      f16

    _Float16* xin   = (_Float16*)(ws + OFF_XIN);
    _Float16* a1    = (_Float16*)(ws + OFF_A1);
    _Float16* a2    = (_Float16*)(ws + OFF_A2);
    _Float16* a3    = (_Float16*)(ws + OFF_A3);
    _Float16* hbuf  = (_Float16*)(ws + OFF_H);
    float*    headF = (float*)   (ws + OFF_HEADF);
    _Float16* zlat  = (_Float16*)(ws + OFF_Z);
    _Float16* d1    = hbuf;
    _Float16* d2    = a3;
    _Float16* e1    = a2;
    _Float16* e2    = a1;
    float*    hbias = (float*)(ws + OFF_HB);

    _Float16* pw = (_Float16*)ws;
    size_t o = 0;
    auto take = [&](size_t n) { _Float16* p = pw + o; o += n; return p; };
    _Float16* pw_c1  = take(32   * 32);
    _Float16* pw_c2  = take(32   * 512);
    _Float16* pw_c3  = take(64   * 288);
    _Float16* pw_fc1 = take(256  * 1024);
    _Float16* pw_hd  = take(48   * 256);
    _Float16* pw_fc2 = take(256  * 32);
    _Float16* pw_fc3 = take(1024 * 256);
    _Float16* pw_d1  = take(32   * 576);
    _Float16* pw_d2  = take(32   * 512);
    _Float16* pw_d3  = take(16   * 512);

    float* xhat   = (float*)d_out;
    float* out_mu = xhat + (size_t)B * 784;
    float* out_ls = out_mu + (size_t)B * 20;

    // ---- pack weights / biases ----
    launch_pack(w_c1,  pw_c1,  32,  32,  16,   32,   1,    4, 4, 0, stream);
    launch_pack(w_c2,  pw_c2,  32,  32,  512,  512,  32,   4, 4, 0, stream);
    launch_pack(w_c3,  pw_c3,  64,  64,  288,  288,  32,   3, 3, 0, stream);
    launch_pack(w_fc1, pw_fc1, 256, 256, 1024, 1024, 1024, 1, 1, 0, stream);
    launch_pack(w_mu,  pw_hd,             20, 20, 256, 256, 256, 1, 1, 0, stream);
    launch_pack(w_ls,  pw_hd + 20 * 256,  20, 28, 256, 256, 256, 1, 1, 0, stream);
    launch_pack(w_fc2, pw_fc2, 256, 256, 20,   32,   20,   1, 1, 0, stream);
    launch_pack(w_fc3, pw_fc3, 1024,1024,256,  256,  256,  1, 1, 0, stream);
    launch_pack(w_d1,  pw_d1,  32,  32,  576,  576,  32,   3, 3, 1, stream);
    launch_pack(w_d2,  pw_d2,  32,  32,  512,  512,  32,   4, 4, 1, stream);
    launch_pack(w_d3,  pw_d3,  1,   16,  512,  512,  1,    4, 4, 1, stream);
    vae_head_bias<<<1, 64, 0, stream>>>(b_mu, b_ls, hbias);

    // ---- input f32 -> f16 ----
    {
        int n = B * 784;
        vae_f32_to_f16<<<(n + 255) / 256, 256, 0, stream>>>(x, xin, n);
    }

    auto grid = [](int M, int Npad) { return dim3((unsigned)((M + 63) / 64), (unsigned)(Npad / 16)); };

    // ---- encoder ----
    vae_wmma_gemm<0,1, 1,28,28, 32,32, 14,14, 4,4, 2,1, 16,32>
        <<<grid(B*196, 32), 128, 0, stream>>>(xin, pw_c1, b_c1, a1, nullptr, B*196);
    vae_wmma_gemm<0,1, 32,14,14, 32,32, 7,7, 4,4, 2,1, 512,512>
        <<<grid(B*49, 32), 128, 0, stream>>>(a1, pw_c2, b_c2, a2, nullptr, B*49);
    vae_wmma_gemm<0,1, 32,7,7, 64,64, 4,4, 3,3, 2,1, 288,288>
        <<<grid(B*16, 64), 128, 0, stream>>>(a2, pw_c3, b_c3, a3, nullptr, B*16);
    vae_wmma_gemm<0,1, 1024,1,1, 256,256, 1,1, 1,1, 1,0, 1024,1024>
        <<<grid(B, 256), 128, 0, stream>>>(a3, pw_fc1, b_fc1, hbuf, nullptr, B);
    vae_wmma_gemm<0,0, 256,1,1, 40,48, 1,1, 1,1, 1,0, 256,256>
        <<<grid(B, 48), 128, 0, stream>>>(hbuf, pw_hd, hbias, nullptr, headF, B);

    // ---- latent recursion + reparameterize ----
    {
        int n = B * 20;
        vae_latent<<<(n + 255) / 256, 256, 0, stream>>>(headF, eps, out_mu, out_ls, zlat, B);
    }

    // ---- decoder ----
    vae_wmma_gemm<0,1, 20,1,1, 256,256, 1,1, 1,1, 1,0, 20,32>
        <<<grid(B, 256), 128, 0, stream>>>(zlat, pw_fc2, b_fc2, d1, nullptr, B);
    vae_wmma_gemm<0,1, 256,1,1, 1024,1024, 1,1, 1,1, 1,0, 256,256>
        <<<grid(B, 1024), 128, 0, stream>>>(d1, pw_fc3, b_fc3, d2, nullptr, B);
    vae_wmma_gemm<1,1, 64,4,4, 32,32, 7,7, 3,3, 2,1, 576,576>
        <<<grid(B*49, 32), 128, 0, stream>>>(d2, pw_d1, b_d1, e1, nullptr, B*49);
    vae_wmma_gemm<1,1, 32,7,7, 32,32, 14,14, 4,4, 2,1, 512,512>
        <<<grid(B*196, 32), 128, 0, stream>>>(e1, pw_d2, b_d2, e2, nullptr, B*196);
    vae_wmma_gemm<1,2, 32,14,14, 1,16, 28,28, 4,4, 2,1, 512,512>
        <<<grid(B*784, 16), 128, 0, stream>>>(e2, pw_d3, b_d3, nullptr, xhat, B*784);
}
